// MultiHeadSelfAttention_71047349010635
// MI455X (gfx1250) — matrix operand order repaired and checked by reading, not compiled, not run on hardware
//
#include <hip/hip_runtime.h>
#include <hip/hip_bf16.h>

// ---------------------------------------------------------------------------
// MHA forward for MI455X (gfx1250): bf16 WMMA path (V_WMMA_F32_16X16X32_BF16)
// with register-prefetch + LDS double buffering.
// ---------------------------------------------------------------------------

#define BATCH     4
#define SEQ       2048
#define D_MODEL   1024
#define NUM_HEADS 16
#define DK        64

typedef __attribute__((ext_vector_type(16))) __bf16 v16bf;
typedef __attribute__((ext_vector_type(8)))  float  v8f;

struct __attribute__((aligned(16))) U4 { unsigned int x, y, z, w; };
struct __attribute__((aligned(16))) F4 { float x, y, z, w; };

union Frag {
    v16bf v;
    U4    q[2];
};

union BV8 { U4 u; __bf16 hh[8]; };

__device__ inline v8f vzero8() {
    v8f z = {0.f, 0.f, 0.f, 0.f, 0.f, 0.f, 0.f, 0.f};
    return z;
}

__device__ inline v8f wmma_bf16(v16bf a, v16bf b, v8f c) {
    // (neg_a, A, neg_b, B, c_mod, C, reuse_a, reuse_b)
    return __builtin_amdgcn_wmma_f32_16x16x32_bf16(false, a, false, b,
                                                   (short)0, c, false, false);
}

// convert 8 f32 -> 8 bf16 and store as one 16B LDS write (v_cvt_pk_bf16_f32 x4)
__device__ inline void cvt_store8(__bf16* dst, F4 a, F4 b) {
    alignas(16) __bf16 t[8] = {(__bf16)a.x, (__bf16)a.y, (__bf16)a.z, (__bf16)a.w,
                               (__bf16)b.x, (__bf16)b.y, (__bf16)b.z, (__bf16)b.w};
    *reinterpret_cast<U4*>(dst) = *reinterpret_cast<const U4*>(t);
}

// ---------------------------------------------------------------------------
// Kernel 1: fused Q/K/V projection + RoPE.
// Grid: (3072/64 n-tiles, 8192/128 m-tiles). Block 256 = 8 waves.
// Each wave computes a 16x64 strip; LDS ping-pong over K-steps of 32.
// ---------------------------------------------------------------------------
__global__ __launch_bounds__(256)
void qkv_rope_kernel(const float* __restrict__ x,
                     const float* __restrict__ wq,
                     const float* __restrict__ wk,
                     const float* __restrict__ wv,
                     __bf16* __restrict__ qws,
                     __bf16* __restrict__ kws,
                     __bf16* __restrict__ vws)
{
    __shared__ __bf16 xs[2][128][32];   // x tile (M x K) bf16
    __shared__ __bf16 wsh[2][64][32];   // w tile (N x K) bf16

    const int tid  = threadIdx.x;
    const int lane = tid & 31;
    const int wave = tid >> 5;
    const int m0   = blockIdx.y * 128;
    const int n0g  = blockIdx.x * 64;     // 0..3071 across {wq,wk,wv}
    const int mat  = n0g >> 10;           // 0=q, 1=k, 2=v
    const int n0   = n0g & (D_MODEL - 1); // column within D
    const float* w = (mat == 0) ? wq : (mat == 1) ? wk : wv;

    const int xrow = tid >> 1, xc0 = (tid & 1) * 16;   // 2 thr/row, 16 cols each
    const int wrow = tid >> 2, wc0 = (tid & 3) * 8;    // 4 thr/row, 8 cols each
    const float* xsrc = x + (size_t)(m0 + xrow) * D_MODEL + xc0;
    const float* wsrc = w + (size_t)(n0 + wrow) * D_MODEL + wc0;

    v8f acc[4];
#pragma unroll
    for (int i = 0; i < 4; ++i) acc[i] = vzero8();

    {   // prologue: stage K-step 0 into buffer 0
        const F4* xp = reinterpret_cast<const F4*>(xsrc);
        cvt_store8(&xs[0][xrow][xc0],     xp[0], xp[1]);
        cvt_store8(&xs[0][xrow][xc0 + 8], xp[2], xp[3]);
        const F4* wp = reinterpret_cast<const F4*>(wsrc);
        cvt_store8(&wsh[0][wrow][wc0], wp[0], wp[1]);
    }
    __syncthreads();

    for (int k0 = 0; k0 < D_MODEL; k0 += 32) {
        const int  cur  = (k0 >> 5) & 1;
        const int  nxt  = cur ^ 1;
        const bool more = (k0 + 32) < D_MODEL;

        F4 xr[4], wr[2];
        if (more) {   // register prefetch of next K-step (overlaps with WMMA)
            const F4* xp = reinterpret_cast<const F4*>(xsrc + k0 + 32);
            xr[0] = xp[0]; xr[1] = xp[1]; xr[2] = xp[2]; xr[3] = xp[3];
            const F4* wp = reinterpret_cast<const F4*>(wsrc + k0 + 32);
            wr[0] = wp[0]; wr[1] = wp[1];
        }

        Frag a;
        {   // A frag: lane<16 -> K {0-7,16-23}; lane>=16 -> K {8-15,24-31}
            int row = (lane & 15) + wave * 16;
            int kb  = (lane < 16) ? 0 : 8;
            a.q[0] = *reinterpret_cast<const U4*>(&xs[cur][row][kb]);
            a.q[1] = *reinterpret_cast<const U4*>(&xs[cur][row][kb + 16]);
        }
        Frag bfr[4];
        const int kb2 = (lane < 16) ? 0 : 8;
#pragma unroll
        for (int sub = 0; sub < 4; ++sub) {
            int col = sub * 16 + (lane & 15);
            bfr[sub].q[0] = *reinterpret_cast<const U4*>(&wsh[cur][col][kb2]);
            bfr[sub].q[1] = *reinterpret_cast<const U4*>(&wsh[cur][col][kb2 + 16]);
        }
#pragma unroll
        for (int sub = 0; sub < 4; ++sub)
            acc[sub] = wmma_bf16(a.v, bfr[sub].v, acc[sub]);

        if (more) {
            cvt_store8(&xs[nxt][xrow][xc0],     xr[0], xr[1]);
            cvt_store8(&xs[nxt][xrow][xc0 + 8], xr[2], xr[3]);
            cvt_store8(&wsh[nxt][wrow][wc0], wr[0], wr[1]);
        }
        __syncthreads();
    }

    // Epilogue: RoPE (q,k only) + scatter to [B,H,S,DK] bf16
    const int halfsel = lane >> 4;
    const int ncol    = lane & 15;
#pragma unroll
    for (int sub = 0; sub < 4; ++sub) {
        int c = n0 + sub * 16 + ncol;       // column in D
        int h = c >> 6;
        int d = c & 63;
        // 10000^(-(d&~1)/64) = exp(-(d&~1) * ln(10000)/64)
        float inv_freq = __expf((float)(d & ~1) * -0.14391156831f);
#pragma unroll
        for (int r = 0; r < 8; ++r) {
            int m  = m0 + wave * 16 + r + 8 * halfsel;
            int s  = m & (SEQ - 1);
            int bb = m >> 11;               // m / SEQ
            float val = acc[sub][r];
            if (mat < 2) {                  // RoPE on q and k
                float other = __shfl_xor(val, 1);
                float ang = (float)s * inv_freq;
                float sn = __sinf(ang);
                float cs = __cosf(ang);
                if ((d & 1) == 0) val = val * cs - other * sn;     // even: x1*c - x2*s
                else              val = other * sn + val * cs;     // odd:  x1*s + x2*c
            }
            size_t idx = ((size_t)(bb * NUM_HEADS + h) * SEQ + s) * DK + d;
            __bf16 o = (__bf16)val;
            if (mat == 0)      qws[idx] = o;
            else if (mat == 1) kws[idx] = o;
            else               vws[idx] = o;
        }
    }
}

// ---------------------------------------------------------------------------
// Kernel 2: causal flash attention.  Grid: (SEQ/128 q-tiles, B*H).
// Block 256 = 8 waves; wave w owns q rows [q0+16w, q0+16w+16).
// K/V tiles double-buffered in LDS with register prefetch.
// ---------------------------------------------------------------------------
__global__ __launch_bounds__(256)
void attn_kernel(const __bf16* __restrict__ qws,
                 const __bf16* __restrict__ kws,
                 const __bf16* __restrict__ vws,
                 __bf16* __restrict__ aout)
{
    __shared__ __bf16 kt[2][32][64];     // key tile [key][d]
    __shared__ __bf16 vt[2][64][32];     // value tile transposed [d][key]
    __shared__ __bf16 pl[8][16][32];     // per-wave P scratch [qrow][key]

    const int tid  = threadIdx.x;
    const int lane = tid & 31;
    const int wave = tid >> 5;
    const int bh   = blockIdx.y;
    const int b    = bh >> 4;
    const int h    = bh & 15;
    const int q0   = blockIdx.x * 128;

    const __bf16* qbase = qws + (size_t)bh * SEQ * DK;
    const __bf16* kbase = kws + (size_t)bh * SEQ * DK;
    const __bf16* vbase = vws + (size_t)bh * SEQ * DK;

    // Load this wave's Q fragments once (two K=32 chunks over d)
    Frag aq[2];
    {
        int row = lane & 15;
        int kb  = (lane < 16) ? 0 : 8;
        const __bf16* qrow = qbase + (size_t)(q0 + wave * 16 + row) * DK;
#pragma unroll
        for (int dc = 0; dc < 2; ++dc) {
            aq[dc].q[0] = *reinterpret_cast<const U4*>(qrow + dc * 32 + kb);
            aq[dc].q[1] = *reinterpret_cast<const U4*>(qrow + dc * 32 + kb + 16);
        }
    }

    v8f   o[4];
    float mrow[8], lrow[8];
#pragma unroll
    for (int i = 0; i < 4; ++i) o[i] = vzero8();
#pragma unroll
    for (int r = 0; r < 8; ++r) { mrow[r] = -1e30f; lrow[r] = 0.0f; }

    const int lkey = tid >> 3;           // 0..31
    const int ld0  = (tid & 7) * 8;      // 8 bf16 per thread

    {   // prologue: stage key tile 0 into buffer 0
        *reinterpret_cast<U4*>(&kt[0][lkey][ld0]) =
            *reinterpret_cast<const U4*>(kbase + (size_t)lkey * DK + ld0);
        BV8 vv;
        vv.u = *reinterpret_cast<const U4*>(vbase + (size_t)lkey * DK + ld0);
#pragma unroll
        for (int j = 0; j < 8; ++j) vt[0][ld0 + j][lkey] = vv.hh[j];
    }
    __syncthreads();

    const int nkt = (q0 >> 5) + 4;       // causal: key tiles of 32
    for (int kti = 0; kti < nkt; ++kti) {
        const int  k0   = kti * 32;
        const int  cur  = kti & 1;
        const int  nxt  = cur ^ 1;
        const bool more = (kti + 1) < nkt;

        U4  kreg;
        BV8 vreg;
        if (more) {   // register prefetch of next key tile
            kreg   = *reinterpret_cast<const U4*>(kbase + (size_t)(k0 + 32 + lkey) * DK + ld0);
            vreg.u = *reinterpret_cast<const U4*>(vbase + (size_t)(k0 + 32 + lkey) * DK + ld0);
        }

        // scores S = Q K^T for 16 q-rows x 32 keys (two 16x16 frags)
        Frag bk[4];
        const int kb2 = (lane < 16) ? 0 : 8;
#pragma unroll
        for (int sub = 0; sub < 2; ++sub) {
            int key = sub * 16 + (lane & 15);
#pragma unroll
            for (int dc = 0; dc < 2; ++dc) {
                bk[sub * 2 + dc].q[0] =
                    *reinterpret_cast<const U4*>(&kt[cur][key][dc * 32 + kb2]);
                bk[sub * 2 + dc].q[1] =
                    *reinterpret_cast<const U4*>(&kt[cur][key][dc * 32 + kb2 + 16]);
            }
        }
        v8f sc[2];
        sc[0] = wmma_bf16(aq[0].v, bk[0].v, vzero8());
        sc[0] = wmma_bf16(aq[1].v, bk[1].v, sc[0]);
        sc[1] = wmma_bf16(aq[0].v, bk[2].v, vzero8());
        sc[1] = wmma_bf16(aq[1].v, bk[3].v, sc[1]);

        // online softmax (row lives across 16 lanes of a half-wave)
        const int qrow_base = q0 + wave * 16 + 8 * (lane >> 4);
        float mnew[8];
#pragma unroll
        for (int r = 0; r < 8; ++r) {
            int qi = qrow_base + r;
#pragma unroll
            for (int sub = 0; sub < 2; ++sub) {
                int key = k0 + sub * 16 + (lane & 15);
                float v = sc[sub][r] * 0.125f;          // 1/sqrt(64)
                if (key > qi) v = -1e30f;               // causal mask
                sc[sub][r] = v;
            }
            float rv = fmaxf(sc[0][r], sc[1][r]);
#pragma unroll
            for (int off = 1; off < 16; off <<= 1)
                rv = fmaxf(rv, __shfl_xor(rv, off));
            mnew[r] = fmaxf(mrow[r], rv);
        }
#pragma unroll
        for (int r = 0; r < 8; ++r) {
            float scale_old = __expf(mrow[r] - mnew[r]);
            int   prow = r + 8 * (lane >> 4);
            float rsum = 0.0f;
#pragma unroll
            for (int sub = 0; sub < 2; ++sub) {
                float v = sc[sub][r];
                float p = (v > -1e29f) ? __expf(v - mnew[r]) : 0.0f;
                rsum += p;
                pl[wave][prow][sub * 16 + (lane & 15)] = (__bf16)p;
            }
#pragma unroll
            for (int off = 1; off < 16; off <<= 1)
                rsum += __shfl_xor(rsum, off);
            lrow[r] = lrow[r] * scale_old + rsum;
            mrow[r] = mnew[r];
#pragma unroll
            for (int j = 0; j < 4; ++j) o[j][r] *= scale_old;
        }

        asm volatile("s_wait_dscnt 0" ::: "memory");   // P stores -> A-frag loads

        Frag ap;
        {
            int row = lane & 15;
            int kb  = (lane < 16) ? 0 : 8;
            ap.q[0] = *reinterpret_cast<const U4*>(&pl[wave][row][kb]);
            ap.q[1] = *reinterpret_cast<const U4*>(&pl[wave][row][kb + 16]);
        }
        Frag bv[4];
#pragma unroll
        for (int j = 0; j < 4; ++j) {
            int d = j * 16 + (lane & 15);
            bv[j].q[0] = *reinterpret_cast<const U4*>(&vt[cur][d][kb2]);
            bv[j].q[1] = *reinterpret_cast<const U4*>(&vt[cur][d][kb2 + 16]);
        }
#pragma unroll
        for (int j = 0; j < 4; ++j)     // O += P @ V  (K=32 keys, 4 d-subtiles)
            o[j] = wmma_bf16(ap.v, bv[j].v, o[j]);

        if (more) {   // stage prefetched tile into alternate buffer
            *reinterpret_cast<U4*>(&kt[nxt][lkey][ld0]) = kreg;
#pragma unroll
            for (int j = 0; j < 8; ++j) vt[nxt][ld0 + j][lkey] = vreg.hh[j];
        }
        __syncthreads();
    }

    // Epilogue: normalize and store to [B,S,D] bf16 for the output projection
    float inv_l[8];
#pragma unroll
    for (int r = 0; r < 8; ++r) inv_l[r] = 1.0f / lrow[r];
    const int srow_base = q0 + wave * 16 + 8 * (lane >> 4);
#pragma unroll
    for (int j = 0; j < 4; ++j) {
        int d = j * 16 + (lane & 15);
#pragma unroll
        for (int r = 0; r < 8; ++r) {
            int s = srow_base + r;
            float v = o[j][r] * inv_l[r];
            aout[((size_t)(b * SEQ + s)) * D_MODEL + h * DK + d] = (__bf16)v;
        }
    }
}

// ---------------------------------------------------------------------------
// Kernel 3: output projection out = attn @ wo^T  (f32 result)
// Grid: (1024/64 n-tiles, 8192/128 m-tiles); LDS ping-pong as kernel 1.
// ---------------------------------------------------------------------------
__global__ __launch_bounds__(256)
void proj_kernel(const __bf16* __restrict__ A,
                 const float*  __restrict__ wo,
                 float* __restrict__ out)
{
    __shared__ __bf16 xs[2][128][32];
    __shared__ __bf16 wsh[2][64][32];

    const int tid  = threadIdx.x;
    const int lane = tid & 31;
    const int wave = tid >> 5;
    const int m0   = blockIdx.y * 128;
    const int n0   = blockIdx.x * 64;

    const int xrow = tid >> 1, xc0 = (tid & 1) * 16;
    const int wrow = tid >> 2, wc0 = (tid & 3) * 8;
    const __bf16* asrc = A  + (size_t)(m0 + xrow) * D_MODEL + xc0;
    const float*  wsrc = wo + (size_t)(n0 + wrow) * D_MODEL + wc0;

    v8f acc[4];
#pragma unroll
    for (int i = 0; i < 4; ++i) acc[i] = vzero8();

    {   // prologue: stage K-step 0
        *reinterpret_cast<U4*>(&xs[0][xrow][xc0]) =
            *reinterpret_cast<const U4*>(asrc);
        *reinterpret_cast<U4*>(&xs[0][xrow][xc0 + 8]) =
            *reinterpret_cast<const U4*>(asrc + 8);
        const F4* wp = reinterpret_cast<const F4*>(wsrc);
        cvt_store8(&wsh[0][wrow][wc0], wp[0], wp[1]);
    }
    __syncthreads();

    for (int k0 = 0; k0 < D_MODEL; k0 += 32) {
        const int  cur  = (k0 >> 5) & 1;
        const int  nxt  = cur ^ 1;
        const bool more = (k0 + 32) < D_MODEL;

        U4 ar0, ar1;
        F4 wr[2];
        if (more) {
            ar0 = *reinterpret_cast<const U4*>(asrc + k0 + 32);
            ar1 = *reinterpret_cast<const U4*>(asrc + k0 + 32 + 8);
            const F4* wp = reinterpret_cast<const F4*>(wsrc + k0 + 32);
            wr[0] = wp[0]; wr[1] = wp[1];
        }

        Frag a;
        {
            int row = (lane & 15) + wave * 16;
            int kb  = (lane < 16) ? 0 : 8;
            a.q[0] = *reinterpret_cast<const U4*>(&xs[cur][row][kb]);
            a.q[1] = *reinterpret_cast<const U4*>(&xs[cur][row][kb + 16]);
        }
        Frag bfr[4];
        const int kb2 = (lane < 16) ? 0 : 8;
#pragma unroll
        for (int sub = 0; sub < 4; ++sub) {
            int col = sub * 16 + (lane & 15);
            bfr[sub].q[0] = *reinterpret_cast<const U4*>(&wsh[cur][col][kb2]);
            bfr[sub].q[1] = *reinterpret_cast<const U4*>(&wsh[cur][col][kb2 + 16]);
        }
#pragma unroll
        for (int sub = 0; sub < 4; ++sub)
            acc[sub] = wmma_bf16(a.v, bfr[sub].v, acc[sub]);

        if (more) {
            *reinterpret_cast<U4*>(&xs[nxt][xrow][xc0])     = ar0;
            *reinterpret_cast<U4*>(&xs[nxt][xrow][xc0 + 8]) = ar1;
            cvt_store8(&wsh[nxt][wrow][wc0], wr[0], wr[1]);
        }
        __syncthreads();
    }

    const int halfsel = lane >> 4;
    const int ncol    = lane & 15;
#pragma unroll
    for (int sub = 0; sub < 4; ++sub) {
#pragma unroll
        for (int r = 0; r < 8; ++r) {
            int m = m0 + wave * 16 + r + 8 * halfsel;
            out[(size_t)m * D_MODEL + n0 + sub * 16 + ncol] = acc[sub][r];
        }
    }
}

// ---------------------------------------------------------------------------
extern "C" void kernel_launch(void* const* d_in, const int* in_sizes, int n_in,
                              void* d_out, int out_size, void* d_ws, size_t ws_size,
                              hipStream_t stream) {
    const float* x  = (const float*)d_in[0];
    const float* wq = (const float*)d_in[1];
    const float* wk = (const float*)d_in[2];
    const float* wv = (const float*)d_in[3];
    const float* wo = (const float*)d_in[4];
    float* out = (float*)d_out;

    const size_t per = (size_t)BATCH * NUM_HEADS * SEQ * DK;  // 8M elems
    __bf16* qws = (__bf16*)d_ws;
    __bf16* kws = qws + per;
    __bf16* vws = kws + per;
    __bf16* aws = vws + per;   // [B,S,D] bf16 attention output

    (void)in_sizes; (void)n_in; (void)out_size; (void)ws_size;

    dim3 blk(256);
    qkv_rope_kernel<<<dim3((3 * D_MODEL) / 64, (BATCH * SEQ) / 128), blk, 0, stream>>>(
        x, wq, wk, wv, qws, kws, vws);
    attn_kernel<<<dim3(SEQ / 128, BATCH * NUM_HEADS), blk, 0, stream>>>(
        qws, kws, vws, aws);
    proj_kernel<<<dim3(D_MODEL / 64, (BATCH * SEQ) / 128), blk, 0, stream>>>(
        aws, wo, out);
}
